// ConditionEncoder_214748365418
// MI455X (gfx1250) — compile-verified, matrix operand order checked by reading, not run on hardware
//
#include <hip/hip_runtime.h>
#include <hip/hip_bf16.h>

typedef __attribute__((ext_vector_type(16))) _Float16 v16h;
typedef __attribute__((ext_vector_type(8)))  float    v8f;

#define DEV_INLINE __device__ __forceinline__

DEV_INLINE v8f wmma_f16(v16h a, v16h b, v8f c) {
  // D = A(16x32 f16) * B(32x16 f16) + C(16x16 f32)
  return __builtin_amdgcn_wmma_f32_16x16x32_f16(false, a, false, b, (short)0, c, false, false);
}

DEV_INLINE v8f vzero8() {
  v8f z = {0.f, 0.f, 0.f, 0.f, 0.f, 0.f, 0.f, 0.f};
  return z;
}

// Position of element (row r, col k) inside a fragment-packed 16 x (32*nkc)
// matrix stored as [kchunk][lane][16 halves].  Derived from the CDNA5 ISA
// 16-bit A-matrix 16x32 VGPR layout (CDNA5 ISA 7.12.2):
//   lane = r | (bit3(klocal)<<4);  vgpr v = bits[2:1](klocal) | (bit4<<2);
//   element j = 2v | bit0(klocal)
DEV_INLINE int frag_pos(int r, int k) {
  int kc = k >> 5, kl = k & 31;
  int lane = r | (((kl >> 3) & 1) << 4);
  int v = ((kl >> 1) & 3) | ((kl >> 4) << 2);
  int j = (v << 1) | (kl & 1);
  return kc * 512 + lane * 16 + j;
}

// ---------------------------------------------------------------------------
// Pre-pack a row-major f32 weight [fin, fout] into fragment-major f16 with the
// K dimension zero-padded to a multiple of 32.  Tile order: (ntile, kchunk).
// ---------------------------------------------------------------------------
__global__ __launch_bounds__(256) void pack_weight_kernel(
    const float* __restrict__ W, _Float16* __restrict__ P,
    int fin, int fout, int total) {
  int e = blockIdx.x * 256 + threadIdx.x;
  if (e >= total) return;
  int k = e / fout, n = e % fout;
  int Kp = total / fout;
  int nkc = Kp >> 5;
  int nt = n >> 4, nl = n & 15;
  int kc = k >> 5, kl = k & 31;
  int lane = nl | (((kl >> 3) & 1) << 4);
  int v = ((kl >> 1) & 3) | ((kl >> 4) << 2);
  int j = (v << 1) | (kl & 1);
  float val = (k < fin) ? W[(size_t)k * fout + n] : 0.f;
  P[((size_t)(nt * nkc + kc)) * 512 + lane * 16 + j] = (_Float16)val;
}

// ---------------------------------------------------------------------------
// Farthest point sampling.  One block per graph; each thread keeps NLOC points
// and their running min-distance entirely in registers.
// ---------------------------------------------------------------------------
template <int NLOC>
__global__ __launch_bounds__(256) void fps_kernel(
    const float* __restrict__ pos, int Np, int M,
    int* __restrict__ idx_out, float* __restrict__ pos_out) {
  int b = blockIdx.x, t = threadIdx.x;
  const float* P = pos + (size_t)b * Np * 3;
  float lx[NLOC], ly[NLOC], lz[NLOC], ld[NLOC];
  int base = t * NLOC;
#pragma unroll
  for (int j = 0; j < NLOC; ++j) {
    int i = base + j;
    lx[j] = P[i * 3 + 0];
    ly[j] = P[i * 3 + 1];
    lz[j] = P[i * 3 + 2];
  }
  __shared__ float sx, sy, sz;
  __shared__ float rv[8];
  __shared__ int ri[8];
  if (t == 0) {
    sx = lx[0]; sy = ly[0]; sz = lz[0];
    idx_out[(size_t)b * M] = 0;
    pos_out[((size_t)b * M) * 3 + 0] = lx[0];
    pos_out[((size_t)b * M) * 3 + 1] = ly[0];
    pos_out[((size_t)b * M) * 3 + 2] = lz[0];
  }
  __syncthreads();
#pragma unroll
  for (int j = 0; j < NLOC; ++j) {
    float dx = lx[j] - sx, dy = ly[j] - sy, dz = lz[j] - sz;
    ld[j] = dx * dx + dy * dy + dz * dz;
  }
  for (int it = 1; it < M; ++it) {
    float bv = -1.f;
    int bi = 0;
#pragma unroll
    for (int j = 0; j < NLOC; ++j)
      if (ld[j] > bv) { bv = ld[j]; bi = base + j; }   // strict > : first max
#pragma unroll
    for (int o = 16; o > 0; o >>= 1) {
      float ov = __shfl_xor(bv, o);
      int oi = __shfl_xor(bi, o);
      if (ov > bv || (ov == bv && oi < bi)) { bv = ov; bi = oi; }
    }
    if ((t & 31) == 0) { rv[t >> 5] = bv; ri[t >> 5] = bi; }
    __syncthreads();
    float gv = rv[0];
    int gi = ri[0];
#pragma unroll
    for (int w = 1; w < 8; ++w)
      if (rv[w] > gv || (rv[w] == gv && ri[w] < gi)) { gv = rv[w]; gi = ri[w]; }
    if (gi >= base && gi < base + NLOC) {  // owner broadcasts chosen point
      int j = gi - base;
      sx = lx[j]; sy = ly[j]; sz = lz[j];
    }
    if (t == 0) idx_out[(size_t)b * M + it] = gi;
    __syncthreads();
    if (t == 0) {
      pos_out[((size_t)b * M + it) * 3 + 0] = sx;
      pos_out[((size_t)b * M + it) * 3 + 1] = sy;
      pos_out[((size_t)b * M + it) * 3 + 2] = sz;
    }
#pragma unroll
    for (int j = 0; j < NLOC; ++j) {
      float dx = lx[j] - sx, dy = ly[j] - sy, dz = lz[j] - sz;
      ld[j] = fminf(ld[j], dx * dx + dy * dy + dz * dz);
    }
  }
}

// ---------------------------------------------------------------------------
// kNN(16): one thread per query, sorted top-16 insertion list in registers,
// source points staged through LDS in chunks of 512.
// ---------------------------------------------------------------------------
__global__ __launch_bounds__(128) void knn_kernel(
    const float* __restrict__ pos, const float* __restrict__ posq,
    int Np, int M, int* __restrict__ nbr) {
  int bpg = M >> 7;  // blocks per graph (M / 128)
  int b = blockIdx.x / bpg;
  int m = (blockIdx.x % bpg) * 128 + threadIdx.x;
  size_t q = (size_t)b * M + m;
  float qx = posq[q * 3 + 0], qy = posq[q * 3 + 1], qz = posq[q * 3 + 2];
  __shared__ float sx[512], sy[512], sz[512];
  float bd[16];
  int bi[16];
#pragma unroll
  for (int k = 0; k < 16; ++k) { bd[k] = __builtin_inff(); bi[k] = 0; }
  for (int c0 = 0; c0 < Np; c0 += 512) {
    __syncthreads();
    for (int i = threadIdx.x; i < 512; i += 128) {
      size_t s = ((size_t)b * Np + c0 + i) * 3;
      sx[i] = pos[s]; sy[i] = pos[s + 1]; sz[i] = pos[s + 2];
    }
    __syncthreads();
    for (int i = 0; i < 512; ++i) {
      float dx = qx - sx[i], dy = qy - sy[i], dz = qz - sz[i];
      float d = dx * dx + dy * dy + dz * dz;
      int idx = c0 + i;
      if (d < bd[15]) {  // strict < keeps earlier index on exact ties (stable)
#pragma unroll
        for (int p = 15; p > 0; --p) {
          if (d < bd[p]) {
            if (d < bd[p - 1]) { bd[p] = bd[p - 1]; bi[p] = bi[p - 1]; }
            else               { bd[p] = d;         bi[p] = idx; }
          }
        }
        if (d < bd[0]) { bd[0] = d; bi[0] = idx; }
      }
    }
  }
#pragma unroll
  for (int k = 0; k < 16; ++k) nbr[q * 16 + k] = bi[k];
}

// ---------------------------------------------------------------------------
// Set-abstraction ResMLP via WMMA.
//   Neighbor pass (SELF=false): 1 wave per query; the query's 16 neighbors are
//     one 16-row tile. relu(A W1 + b1) -> hidden; hidden W2 + A Ws + b2 + bs
//     -> relu -> self-loop mask -> column max -> h_out[b, m, :].
//   Self pass (SELF=true): 1 wave per 16 queries; rows are self features
//     (x[:M], pos[:M]-pos_i); output max-combined into h_out.
// ---------------------------------------------------------------------------
template <int C, int CIN, int KP, int H, int COUT, bool SELF>
__global__ __launch_bounds__(32) void sa_mlp_kernel(
    const float* __restrict__ h_in, const float* __restrict__ pos_in,
    const float* __restrict__ pos_q, const int* __restrict__ nbr,
    float* __restrict__ h_out,
    const _Float16* __restrict__ pW1, const _Float16* __restrict__ pW2,
    const _Float16* __restrict__ pWs,
    const float* __restrict__ b1v, const float* __restrict__ b2v,
    const float* __restrict__ bsv, int M, int NP) {
  constexpr int NKA = KP / 32;  // K chunks of the input features
  constexpr int NKH = H / 32;   // K chunks of the hidden features
  __shared__ __align__(32) _Float16 sA[16 * KP];
  __shared__ __align__(32) _Float16 sH[16 * H];
  __shared__ int sN[16];
  const int lane = threadIdx.x;
  const int g = blockIdx.x;
  int b, mq, m0;
  if (!SELF) {
    b = g / M; mq = g - b * M; m0 = 0;
    if (lane < 16) sN[lane] = nbr[((size_t)(b * M + mq)) * 16 + lane];
    __syncthreads();
  } else {
    int bpg = M >> 4;
    b = g / bpg; m0 = (g - b * bpg) << 4; mq = 0;
  }

  // Build the 16 x KP feature tile directly in fragment-packed layout.
  for (int e = lane; e < 16 * KP; e += 32) {
    int r = e & 15, k = e >> 4;
    float val = 0.f;
    if (!SELF) {
      if (k < CIN) {
        int src = sN[r];
        if (k < C) {
          val = h_in[((size_t)b * NP + src) * C + k];
        } else {
          int d = k - C;
          val = pos_in[((size_t)b * NP + src) * 3 + d] -
                pos_q[((size_t)(b * M + mq)) * 3 + d];
        }
      }
    } else {
      int row = m0 + r;
      if (k < C) {
        val = h_in[((size_t)b * NP + row) * C + k];
      } else if (k < CIN) {
        int d = k - C;
        val = pos_in[((size_t)b * NP + row) * 3 + d] -
              pos_q[((size_t)(b * M + row)) * 3 + d];
      }
    }
    sA[frag_pos(r, k)] = (_Float16)val;
  }
  __syncthreads();

  const int nl = lane & 15, hi = lane >> 4;

  // GEMM 1: hidden = relu(A * W1 + b1), repacked as f16 fragments in LDS.
  for (int nt = 0; nt < H / 16; ++nt) {
    v8f acc = vzero8();
#pragma unroll
    for (int kc = 0; kc < NKA; ++kc) {
      v16h a = *(const v16h*)(sA + kc * 512 + lane * 16);
      v16h w = *(const v16h*)(pW1 + ((size_t)(nt * NKA + kc)) * 512 + lane * 16);
      acc = wmma_f16(a, w, acc);
    }
    float bias = b1v[nt * 16 + nl];
    int kcol = nt * 16 + nl;
#pragma unroll
    for (int v = 0; v < 8; ++v) {
      int m = v + 8 * hi;  // D layout: lane n = lane&15, rows v + 8*(lane>=16)
      sH[frag_pos(m, kcol)] = (_Float16)fmaxf(acc[v] + bias, 0.f);
    }
  }
  __syncthreads();

  unsigned vm = 0xFFFFu;  // valid-row mask (drop numeric self loops)
  if (!SELF) {
    vm = 0;
#pragma unroll
    for (int r = 0; r < 16; ++r)
      if (sN[r] != mq) vm |= (1u << r);
  }

  // GEMM 2: out = relu(hidden * W2 + A * Ws + b2 + bs); reduce / combine.
  for (int nt = 0; nt < COUT / 16; ++nt) {
    v8f acc = vzero8();
#pragma unroll
    for (int kc = 0; kc < NKH; ++kc) {
      v16h a = *(const v16h*)(sH + kc * 512 + lane * 16);
      v16h w = *(const v16h*)(pW2 + ((size_t)(nt * NKH + kc)) * 512 + lane * 16);
      acc = wmma_f16(a, w, acc);
    }
#pragma unroll
    for (int kc = 0; kc < NKA; ++kc) {
      v16h a = *(const v16h*)(sA + kc * 512 + lane * 16);
      v16h w = *(const v16h*)(pWs + ((size_t)(nt * NKA + kc)) * 512 + lane * 16);
      acc = wmma_f16(a, w, acc);
    }
    int n = nt * 16 + nl;
    float bias = b2v[n] + bsv[n];
    if (!SELF) {
      float mx = -__builtin_inff();
#pragma unroll
      for (int v = 0; v < 8; ++v) {
        int m = v + 8 * hi;
        float x = fmaxf(acc[v] + bias, 0.f);
        if ((vm >> m) & 1) mx = fmaxf(mx, x);
      }
      mx = fmaxf(mx, __shfl_xor(mx, 16));  // combine row halves (same column)
      if (lane < 16) h_out[((size_t)(b * M + mq)) * COUT + n] = mx;
    } else {
#pragma unroll
      for (int v = 0; v < 8; ++v) {
        int m = v + 8 * hi;
        float x = fmaxf(acc[v] + bias, 0.f);
        float* p = h_out + ((size_t)(b * M + m0 + m)) * COUT + n;
        *p = fmaxf(*p, x);  // each element touched by exactly one lane
      }
    }
  }
}

// ---------------------------------------------------------------------------
// Global max pool over the 256 remaining points -> [8, 512]
// ---------------------------------------------------------------------------
__global__ __launch_bounds__(512) void pool_kernel(const float* __restrict__ h4,
                                                   float* __restrict__ gp) {
  int b = blockIdx.x, c = threadIdx.x;
  float mx = -__builtin_inff();
  for (int r = 0; r < 256; ++r)
    mx = fmaxf(mx, h4[((size_t)b * 256 + r) * 512 + c]);
  gp[(size_t)b * 512 + c] = mx;
}

// ---------------------------------------------------------------------------
// Head: u = relu(resmlp_512(g)); out = u @ W3 + b3.  Rows padded 8 -> 16 so
// a single wave runs the whole 3-GEMM chain with WMMA.
// ---------------------------------------------------------------------------
__global__ __launch_bounds__(32) void head_kernel(
    const float* __restrict__ gp, float* __restrict__ out,
    const _Float16* __restrict__ pW1, const _Float16* __restrict__ pW2,
    const _Float16* __restrict__ pWs, const _Float16* __restrict__ pW3,
    const float* __restrict__ b1v, const float* __restrict__ b2v,
    const float* __restrict__ bsv, const float* __restrict__ b3v) {
  __shared__ __align__(32) _Float16 sA[16 * 512];
  __shared__ __align__(32) _Float16 sT[16 * 512];
  __shared__ __align__(32) _Float16 sU[16 * 512];
  const int lane = threadIdx.x;
  const int nl = lane & 15, hi = lane >> 4;
  for (int e = lane; e < 16 * 512; e += 32) {
    int r = e & 15, k = e >> 4;
    float val = (r < 8) ? gp[(size_t)r * 512 + k] : 0.f;
    sA[frag_pos(r, k)] = (_Float16)val;
  }
  __syncthreads();
  for (int nt = 0; nt < 32; ++nt) {  // T = relu(A W1 + b1)
    v8f acc = vzero8();
#pragma unroll
    for (int kc = 0; kc < 16; ++kc)
      acc = wmma_f16(*(const v16h*)(sA + kc * 512 + lane * 16),
                     *(const v16h*)(pW1 + ((size_t)(nt * 16 + kc)) * 512 + lane * 16), acc);
    float bias = b1v[nt * 16 + nl];
#pragma unroll
    for (int v = 0; v < 8; ++v)
      sT[frag_pos(v + 8 * hi, nt * 16 + nl)] = (_Float16)fmaxf(acc[v] + bias, 0.f);
  }
  __syncthreads();
  for (int nt = 0; nt < 32; ++nt) {  // U = relu(T W2 + A Ws + b2 + bs)
    v8f acc = vzero8();
#pragma unroll
    for (int kc = 0; kc < 16; ++kc)
      acc = wmma_f16(*(const v16h*)(sT + kc * 512 + lane * 16),
                     *(const v16h*)(pW2 + ((size_t)(nt * 16 + kc)) * 512 + lane * 16), acc);
#pragma unroll
    for (int kc = 0; kc < 16; ++kc)
      acc = wmma_f16(*(const v16h*)(sA + kc * 512 + lane * 16),
                     *(const v16h*)(pWs + ((size_t)(nt * 16 + kc)) * 512 + lane * 16), acc);
    float bias = b2v[nt * 16 + nl] + bsv[nt * 16 + nl];
#pragma unroll
    for (int v = 0; v < 8; ++v)
      sU[frag_pos(v + 8 * hi, nt * 16 + nl)] = (_Float16)fmaxf(acc[v] + bias, 0.f);
  }
  __syncthreads();
  for (int nt = 0; nt < 16; ++nt) {  // out = U W3 + b3 (rows 0..7 only)
    v8f acc = vzero8();
#pragma unroll
    for (int kc = 0; kc < 16; ++kc)
      acc = wmma_f16(*(const v16h*)(sU + kc * 512 + lane * 16),
                     *(const v16h*)(pW3 + ((size_t)(nt * 16 + kc)) * 512 + lane * 16), acc);
    float bias = b3v[nt * 16 + nl];
#pragma unroll
    for (int v = 0; v < 8; ++v) {
      int m = v + 8 * hi;
      if (m < 8) out[(size_t)m * 256 + nt * 16 + nl] = acc[v] + bias;
    }
  }
}

// ---------------------------------------------------------------------------
// Launcher
// ---------------------------------------------------------------------------
extern "C" void kernel_launch(void* const* d_in, const int* in_sizes, int n_in,
                              void* d_out, int out_size, void* d_ws, size_t ws_size,
                              hipStream_t stream) {
  (void)in_sizes; (void)n_in; (void)out_size; (void)ws_size;
  // Input order: x, pos, batch, then params flattened with JAX-sorted dict
  // keys: g{W1,W2,W3,Ws,b1,b2,b3,bs}, sa1..sa4{W1,W2,Ws,b1,b2,bs}
  const float* x   = (const float*)d_in[0];
  const float* pos = (const float*)d_in[1];

  char* w = (char*)d_ws;
  size_t off = 0;
  auto carve = [&](size_t bytes) -> void* {
    void* p = (void*)(w + off);
    off += (bytes + 255) & ~(size_t)255;
    return p;
  };
  float* pos1 = (float*)carve(8ull * 2048 * 3 * 4);
  float* pos2 = (float*)carve(8ull * 1024 * 3 * 4);
  float* pos3 = (float*)carve(8ull * 512 * 3 * 4);
  float* pos4 = (float*)carve(8ull * 256 * 3 * 4);
  float* h1 = (float*)carve(8ull * 2048 * 64 * 4);
  float* h2 = (float*)carve(8ull * 1024 * 128 * 4);
  float* h3 = (float*)carve(8ull * 512 * 256 * 4);
  float* h4 = (float*)carve(8ull * 256 * 512 * 4);
  int* idx1 = (int*)carve(8ull * 2048 * 4);
  int* idx2 = (int*)carve(8ull * 1024 * 4);
  int* idx3 = (int*)carve(8ull * 512 * 4);
  int* idx4 = (int*)carve(8ull * 256 * 4);
  int* nbr1 = (int*)carve(8ull * 2048 * 16 * 4);
  int* nbr2 = (int*)carve(8ull * 1024 * 16 * 4);
  int* nbr3 = (int*)carve(8ull * 512 * 16 * 4);
  int* nbr4 = (int*)carve(8ull * 256 * 16 * 4);
  float* gpool = (float*)carve(8ull * 512 * 4);
  _Float16* P_sa1W1 = (_Float16*)carve(32ull * 32 * 2);
  _Float16* P_sa1W2 = (_Float16*)carve(32ull * 64 * 2);
  _Float16* P_sa1Ws = (_Float16*)carve(32ull * 64 * 2);
  _Float16* P_sa2W1 = (_Float16*)carve(96ull * 64 * 2);
  _Float16* P_sa2W2 = (_Float16*)carve(64ull * 128 * 2);
  _Float16* P_sa2Ws = (_Float16*)carve(96ull * 128 * 2);
  _Float16* P_sa3W1 = (_Float16*)carve(160ull * 128 * 2);
  _Float16* P_sa3W2 = (_Float16*)carve(128ull * 256 * 2);
  _Float16* P_sa3Ws = (_Float16*)carve(160ull * 256 * 2);
  _Float16* P_sa4W1 = (_Float16*)carve(288ull * 256 * 2);
  _Float16* P_sa4W2 = (_Float16*)carve(256ull * 512 * 2);
  _Float16* P_sa4Ws = (_Float16*)carve(288ull * 512 * 2);
  _Float16* P_gW1 = (_Float16*)carve(512ull * 512 * 2);
  _Float16* P_gW2 = (_Float16*)carve(512ull * 512 * 2);
  _Float16* P_gWs = (_Float16*)carve(512ull * 512 * 2);
  _Float16* P_gW3 = (_Float16*)carve(512ull * 256 * 2);

  auto pack = [&](const void* W, _Float16* P, int fin, int fout) {
    int Kp = (fin + 31) & ~31;
    int total = Kp * fout;
    pack_weight_kernel<<<(total + 255) / 256, 256, 0, stream>>>(
        (const float*)W, P, fin, fout, total);
  };
  pack(d_in[11], P_sa1W1, 6, 32);
  pack(d_in[12], P_sa1W2, 32, 64);
  pack(d_in[13], P_sa1Ws, 6, 64);
  pack(d_in[17], P_sa2W1, 67, 64);
  pack(d_in[18], P_sa2W2, 64, 128);
  pack(d_in[19], P_sa2Ws, 67, 128);
  pack(d_in[23], P_sa3W1, 131, 128);
  pack(d_in[24], P_sa3W2, 128, 256);
  pack(d_in[25], P_sa3Ws, 131, 256);
  pack(d_in[29], P_sa4W1, 259, 256);
  pack(d_in[30], P_sa4W2, 256, 512);
  pack(d_in[31], P_sa4Ws, 259, 512);
  pack(d_in[3], P_gW1, 512, 512);
  pack(d_in[4], P_gW2, 512, 512);
  pack(d_in[6], P_gWs, 512, 512);
  pack(d_in[5], P_gW3, 512, 256);

  // ---- Level 1: 4096 -> 2048, C=3, MLP 6->32->64 ----
  fps_kernel<16><<<8, 256, 0, stream>>>(pos, 4096, 2048, idx1, pos1);
  knn_kernel<<<8 * (2048 / 128), 128, 0, stream>>>(pos, pos1, 4096, 2048, nbr1);
  sa_mlp_kernel<3, 6, 32, 32, 64, false><<<8 * 2048, 32, 0, stream>>>(
      x, pos, pos1, nbr1, h1, P_sa1W1, P_sa1W2, P_sa1Ws,
      (const float*)d_in[14], (const float*)d_in[15], (const float*)d_in[16], 2048, 4096);
  sa_mlp_kernel<3, 6, 32, 32, 64, true><<<8 * 2048 / 16, 32, 0, stream>>>(
      x, pos, pos1, nbr1, h1, P_sa1W1, P_sa1W2, P_sa1Ws,
      (const float*)d_in[14], (const float*)d_in[15], (const float*)d_in[16], 2048, 4096);

  // ---- Level 2: 2048 -> 1024, C=64, MLP 67->64->128 ----
  fps_kernel<8><<<8, 256, 0, stream>>>(pos1, 2048, 1024, idx2, pos2);
  knn_kernel<<<8 * (1024 / 128), 128, 0, stream>>>(pos1, pos2, 2048, 1024, nbr2);
  sa_mlp_kernel<64, 67, 96, 64, 128, false><<<8 * 1024, 32, 0, stream>>>(
      h1, pos1, pos2, nbr2, h2, P_sa2W1, P_sa2W2, P_sa2Ws,
      (const float*)d_in[20], (const float*)d_in[21], (const float*)d_in[22], 1024, 2048);
  sa_mlp_kernel<64, 67, 96, 64, 128, true><<<8 * 1024 / 16, 32, 0, stream>>>(
      h1, pos1, pos2, nbr2, h2, P_sa2W1, P_sa2W2, P_sa2Ws,
      (const float*)d_in[20], (const float*)d_in[21], (const float*)d_in[22], 1024, 2048);

  // ---- Level 3: 1024 -> 512, C=128, MLP 131->128->256 ----
  fps_kernel<4><<<8, 256, 0, stream>>>(pos2, 1024, 512, idx3, pos3);
  knn_kernel<<<8 * (512 / 128), 128, 0, stream>>>(pos2, pos3, 1024, 512, nbr3);
  sa_mlp_kernel<128, 131, 160, 128, 256, false><<<8 * 512, 32, 0, stream>>>(
      h2, pos2, pos3, nbr3, h3, P_sa3W1, P_sa3W2, P_sa3Ws,
      (const float*)d_in[26], (const float*)d_in[27], (const float*)d_in[28], 512, 1024);
  sa_mlp_kernel<128, 131, 160, 128, 256, true><<<8 * 512 / 16, 32, 0, stream>>>(
      h2, pos2, pos3, nbr3, h3, P_sa3W1, P_sa3W2, P_sa3Ws,
      (const float*)d_in[26], (const float*)d_in[27], (const float*)d_in[28], 512, 1024);

  // ---- Level 4: 512 -> 256, C=256, MLP 259->256->512 ----
  fps_kernel<2><<<8, 256, 0, stream>>>(pos3, 512, 256, idx4, pos4);
  knn_kernel<<<8 * (256 / 128), 128, 0, stream>>>(pos3, pos4, 512, 256, nbr4);
  sa_mlp_kernel<256, 259, 288, 256, 512, false><<<8 * 256, 32, 0, stream>>>(
      h3, pos3, pos4, nbr4, h4, P_sa4W1, P_sa4W2, P_sa4Ws,
      (const float*)d_in[32], (const float*)d_in[33], (const float*)d_in[34], 256, 512);
  sa_mlp_kernel<256, 259, 288, 256, 512, true><<<8 * 256 / 16, 32, 0, stream>>>(
      h3, pos3, pos4, nbr4, h4, P_sa4W1, P_sa4W2, P_sa4Ws,
      (const float*)d_in[32], (const float*)d_in[33], (const float*)d_in[34], 256, 512);

  // ---- Global pool + head ----
  pool_kernel<<<8, 512, 0, stream>>>(h4, gpool);
  head_kernel<<<1, 32, 0, stream>>>(
      gpool, (float*)d_out, P_gW1, P_gW2, P_gWs, P_gW3,
      (const float*)d_in[7], (const float*)d_in[8],
      (const float*)d_in[10], (const float*)d_in[9]);
}
